// Attention_72670846649042
// MI455X (gfx1250) — matrix operand order repaired and checked by reading, not compiled, not run on hardware
//
#include <hip/hip_runtime.h>
#include <hip/hip_bf16.h>
#include <math.h>

typedef __attribute__((ext_vector_type(16))) _Float16 v16h;
typedef __attribute__((ext_vector_type(8)))  _Float16 v8h;
typedef __attribute__((ext_vector_type(8)))  float    v8f;

#define BB   512
#define LL   25
#define EE   300
#define EPAD 320
#define HH   1024
#define DD   256
#define TT   128
#define CAT2 1280
#define G3H  3072

// ---------------------------------------------------------------------------
// Generic f32 -> f16 convert with zero-padding of the inner dimension.
// ---------------------------------------------------------------------------
__global__ __launch_bounds__(256)
void cvt_f16_pad(const float* __restrict__ src, _Float16* __restrict__ dst,
                 int rows, int cin, int cout) {
  long i = (long)blockIdx.x * blockDim.x + threadIdx.x;
  long total = (long)rows * cout;
  if (i >= total) return;
  int r = (int)(i / cout), c = (int)(i % cout);
  dst[i] = (c < cin) ? (_Float16)src[(long)r * cin + c] : (_Float16)0.0f;
}

// ---------------------------------------------------------------------------
// Encoder embedding: gather w2v[tok], renorm to max L2 = 10, f16 + pad to 320.
// One block per (b,l) row.
// ---------------------------------------------------------------------------
__global__ __launch_bounds__(256)
void enc_embed(const int* __restrict__ tokens, const float* __restrict__ w2v,
               _Float16* __restrict__ x) {
  int row = blockIdx.x;                       // 0..B*L-1
  const float* src = w2v + (long)tokens[row] * EE;
  float ss = 0.f;
  for (int c = threadIdx.x; c < EE; c += 256) { float v = src[c]; ss += v * v; }
  __shared__ float red[256];
  red[threadIdx.x] = ss; __syncthreads();
  for (int s = 128; s > 0; s >>= 1) {
    if (threadIdx.x < s) red[threadIdx.x] += red[threadIdx.x + s];
    __syncthreads();
  }
  float n  = sqrtf(red[0]);
  float sc = fminf(1.0f, 10.0f / (n + 1e-7f));
  _Float16* dst = x + (long)row * EPAD;
  for (int c = threadIdx.x; c < EPAD; c += 256)
    dst[c] = (c < EE) ? (_Float16)(src[c] * sc) : (_Float16)0.0f;
}

// ---------------------------------------------------------------------------
// Decoder embedding: gather dec_emb[inp], renorm max L2 = 1, write f32 copy
// (for attention scores) and f16 into cat2[:, 0:256].
// ---------------------------------------------------------------------------
__global__ __launch_bounds__(256)
void dec_embed(const int* __restrict__ inp, const float* __restrict__ dec_emb,
               float* __restrict__ emb_f32, _Float16* __restrict__ cat2) {
  int b = blockIdx.x;
  const float* src = dec_emb + (long)inp[b] * DD;
  float v = src[threadIdx.x];
  __shared__ float red[256];
  red[threadIdx.x] = v * v; __syncthreads();
  for (int s = 128; s > 0; s >>= 1) {
    if (threadIdx.x < s) red[threadIdx.x] += red[threadIdx.x + s];
    __syncthreads();
  }
  float sc = fminf(1.0f, 1.0f / (sqrtf(red[0]) + 1e-7f));
  float o  = v * sc;
  emb_f32[(long)b * DD + threadIdx.x]   = o;
  cat2[(long)b * CAT2 + threadIdx.x]    = (_Float16)o;
}

// ---------------------------------------------------------------------------
// WMMA GEMM: C[M,N] = A[M,K] * W[N,K]^T + bias, optional BN+ReLU epilogue,
// optional strided f32 and/or f16 outputs.
// Block = 256 threads = 8 waves; wave tile 32x32 (2x2 v_wmma_f32_16x16x32_f16);
// block tile 128(M) x 64(N). Assumes M%128==0, N%64==0, K%32==0 (true here).
// Fragment layouts (contiguous k = 16*(lane>>4) + s in both A and B, not the ISA table order):
//   A (16x32 f16): lane m=lane&15, kb=16*(lane>>4): halves [kb,kb+8) then [kb+8,kb+16)
//   B (32x16 f16): lane n=lane&15, kb=16*(lane>>4): halves [kb,kb+16) of row n of W
//   C (16x16 f32): VGPR r -> M = r + 8*(lane>>4), N = lane&15
// ---------------------------------------------------------------------------
__global__ __launch_bounds__(256)
void gemm_wmma_f16(const _Float16* __restrict__ A, long lda,
                   const _Float16* __restrict__ W, long ldb,
                   const float* __restrict__ bias,
                   float* __restrict__ C, long ldc,
                   _Float16* __restrict__ Ch, long ldch,
                   int M, int N, int K,
                   const float* __restrict__ bnp, int relu) {
  const int wave = threadIdx.x >> 5;
  const int lane = threadIdx.x & 31;
  const int wm = wave & 3;                 // 4 waves along M
  const int wn = wave >> 2;                // 2 waves along N
  const int m0 = blockIdx.y * 128 + wm * 32;
  const int n0 = blockIdx.x * 64  + wn * 32;
  if (m0 >= M || n0 >= N) return;          // wave-uniform

  const int mA  = lane & 15;
  const int kbA = (lane >> 4) << 4;        // 0 or 16
  const int nB  = lane & 15;
  const int kbB = (lane >> 4) << 4;        // 0 or 16

  v8f acc[2][2] = {};

  const _Float16* a0p = A + (long)(m0 + mA) * lda + kbA;
  const _Float16* a1p = a0p + 16L * lda;
  const _Float16* b0p = W + (long)(n0 + nB) * ldb + kbB;
  const _Float16* b1p = b0p + 16L * ldb;

  for (int k = 0; k < K; k += 32) {
    v16h a0, a1, b0, b1;
    *((v8h*)&a0)     = *(const v8h*)(a0p + k);
    *((v8h*)&a0 + 1) = *(const v8h*)(a0p + k + 8);
    *((v8h*)&a1)     = *(const v8h*)(a1p + k);
    *((v8h*)&a1 + 1) = *(const v8h*)(a1p + k + 8);
    b0 = *(const v16h*)(b0p + k);
    b1 = *(const v16h*)(b1p + k);
    if (k + 32 < K) {                      // stream next weight K-tile
      __builtin_prefetch(b0p + k + 32, 0, 1);
      __builtin_prefetch(b1p + k + 32, 0, 1);
    }
    acc[0][0] = __builtin_amdgcn_wmma_f32_16x16x32_f16(false, a0, false, b0,
                (short)0, acc[0][0], false, false);
    acc[0][1] = __builtin_amdgcn_wmma_f32_16x16x32_f16(false, a0, false, b1,
                (short)0, acc[0][1], false, false);
    acc[1][0] = __builtin_amdgcn_wmma_f32_16x16x32_f16(false, a1, false, b0,
                (short)0, acc[1][0], false, false);
    acc[1][1] = __builtin_amdgcn_wmma_f32_16x16x32_f16(false, a1, false, b1,
                (short)0, acc[1][1], false, false);
  }

  float es = 1.f, eb = 0.f;
  if (bnp) {                                // BatchNorm2d(1) eval-mode affine
    float s = bnp[0] * rsqrtf(bnp[3] + 1e-5f);
    es = s; eb = bnp[1] - bnp[2] * s;
  }

  const int mr = (lane >> 4) << 3;          // 0 or 8
  const int nc = lane & 15;
#pragma unroll
  for (int i = 0; i < 2; ++i)
#pragma unroll
    for (int j = 0; j < 2; ++j) {
      const int gn = n0 + j * 16 + nc;
      const float bv = bias ? bias[gn] : 0.f;
#pragma unroll
      for (int r = 0; r < 8; ++r) {
        const int gm = m0 + i * 16 + mr + r;
        float v = acc[i][j][r] + bv;
        v = v * es + eb;
        if (relu) v = fmaxf(v, 0.f);
        if (C)  C [(long)gm * ldc  + gn] = v;
        if (Ch) Ch[(long)gm * ldch + gn] = (_Float16)v;
      }
    }
}

// ---------------------------------------------------------------------------
// GRU elementwise gate combine. gi/gh are (B, 3H) f32 with biases already
// added by the GEMMs. Updates h in place, writes h_f16, optionally writes
// enc_out slice (encoder) or bn1(h2) as f16 (decoder, feeds output GEMM).
// grid = (H/256, B)
// ---------------------------------------------------------------------------
__global__ __launch_bounds__(256)
void gru_elem(const float* __restrict__ gi, const float* __restrict__ gh,
              float* __restrict__ h, _Float16* __restrict__ h16,
              float* __restrict__ enc_out_t,          // +t*H, stride L*H, or null
              _Float16* __restrict__ bnh16,           // or null
              const float* __restrict__ bn1) {        // or null
  int b = blockIdx.y;
  int j = blockIdx.x * 256 + threadIdx.x;
  const float* gib = gi + (long)b * G3H;
  const float* ghb = gh + (long)b * G3H;
  float ir = gib[j], iz = gib[HH + j], inn = gib[2 * HH + j];
  float hr = ghb[j], hz = ghb[HH + j], hn  = ghb[2 * HH + j];
  float r = 1.f / (1.f + __expf(-(ir + hr)));
  float z = 1.f / (1.f + __expf(-(iz + hz)));
  float n = tanhf(inn + r * hn);
  float hprev = h[(long)b * HH + j];
  float h2 = (1.f - z) * n + z * hprev;
  h[(long)b * HH + j]   = h2;
  h16[(long)b * HH + j] = (_Float16)h2;
  if (enc_out_t) enc_out_t[(long)b * LL * HH + j] = h2;
  if (bnh16) {
    float s  = bn1[0] * rsqrtf(bn1[3] + 1e-5f);
    float sh = bn1[1] - bn1[2] * s;
    bnh16[(long)b * HH + j] = (_Float16)(h2 * s + sh);
  }
}

// ---------------------------------------------------------------------------
// Attention: scores = [emb,h] @ attn_W^T + attn_b -> softmax over L=25 ->
// applied[b,:] = sum_l aw[l] * enc_out[b,l,:], written f16 into cat2[:,256:].
// One block of 256 per batch row. Tiny (K=1280, L=25) -> plain VALU.
// ---------------------------------------------------------------------------
__global__ __launch_bounds__(256)
void attn_kernel(const float* __restrict__ emb, const float* __restrict__ h,
                 const float* __restrict__ attn_W, const float* __restrict__ attn_b,
                 const float* __restrict__ enc_out, _Float16* __restrict__ cat2_app) {
  int b = blockIdx.x;
  __shared__ float sc[LL];
  __shared__ float red[256];
  for (int l = 0; l < LL; ++l) {
    const float* wrow = attn_W + (long)l * CAT2;
    float s = 0.f;
    for (int k = threadIdx.x; k < CAT2; k += 256) {
      float xk = (k < DD) ? emb[(long)b * DD + k] : h[(long)b * HH + (k - DD)];
      s += xk * wrow[k];
    }
    red[threadIdx.x] = s; __syncthreads();
    for (int t = 128; t > 0; t >>= 1) {
      if (threadIdx.x < t) red[threadIdx.x] += red[threadIdx.x + t];
      __syncthreads();
    }
    if (threadIdx.x == 0) sc[l] = red[0] + attn_b[l];
    __syncthreads();
  }
  if (threadIdx.x == 0) {
    float mx = -INFINITY;
    for (int l = 0; l < LL; ++l) mx = fmaxf(mx, sc[l]);
    float se = 0.f;
    for (int l = 0; l < LL; ++l) { sc[l] = __expf(sc[l] - mx); se += sc[l]; }
    float inv = 1.f / se;
    for (int l = 0; l < LL; ++l) sc[l] *= inv;
  }
  __syncthreads();
  const float* eo = enc_out + (long)b * LL * HH;
  for (int j = threadIdx.x; j < HH; j += 256) {
    float acc = 0.f;
    for (int l = 0; l < LL; ++l) acc += sc[l] * eo[(long)l * HH + j];
    cat2_app[(long)b * CAT2 + j] = (_Float16)acc;   // base already +DD offset? no: see launch
  }
}

// ---------------------------------------------------------------------------
// In-place log-softmax over T=128 + greedy argmax (first max wins, like jnp).
// Rows strided by L*T inside d_out. grid=B, block=T.
// ---------------------------------------------------------------------------
__global__ __launch_bounds__(TT)
void lsm_argmax(float* __restrict__ out, long ldo, int* __restrict__ next_inp) {
  int b = blockIdx.x, t = threadIdx.x;
  float* row = out + (long)b * ldo;
  float v = row[t];
  __shared__ float rv[TT];
  __shared__ int   ri[TT];
  rv[t] = v; ri[t] = t; __syncthreads();
  for (int s = TT / 2; s > 0; s >>= 1) {
    if (t < s) {
      bool take = (rv[t + s] > rv[t]) || (rv[t + s] == rv[t] && ri[t + s] < ri[t]);
      if (take) { rv[t] = rv[t + s]; ri[t] = ri[t + s]; }
    }
    __syncthreads();
  }
  float mx = rv[0]; int am = ri[0];
  __syncthreads();
  rv[t] = __expf(v - mx); __syncthreads();
  for (int s = TT / 2; s > 0; s >>= 1) {
    if (t < s) rv[t] += rv[t + s];
    __syncthreads();
  }
  float lse = logf(rv[0]) + mx;
  row[t] = v - lse;
  if (t == 0) next_inp[b] = am;
}

__global__ void fill_int(int* p, int v, int n) {
  int i = blockIdx.x * blockDim.x + threadIdx.x;
  if (i < n) p[i] = v;
}

// ---------------------------------------------------------------------------
extern "C" void kernel_launch(void* const* d_in, const int* in_sizes, int n_in,
                              void* d_out, int out_size, void* d_ws, size_t ws_size,
                              hipStream_t stream) {
  const int*   tokens  = (const int*)  d_in[0];
  const float* w2v     = (const float*)d_in[1];
  const float* encWih  = (const float*)d_in[2];
  const float* encWhh  = (const float*)d_in[3];
  const float* enc_bih = (const float*)d_in[4];
  const float* enc_bhh = (const float*)d_in[5];
  const float* dec_emb = (const float*)d_in[6];
  const float* attn_W  = (const float*)d_in[7];
  const float* attn_b  = (const float*)d_in[8];
  const float* comb_W  = (const float*)d_in[9];
  const float* comb_b  = (const float*)d_in[10];
  const float* decWih  = (const float*)d_in[11];
  const float* decWhh  = (const float*)d_in[12];
  const float* dec_bih = (const float*)d_in[13];
  const float* dec_bhh = (const float*)d_in[14];
  const float* out_W   = (const float*)d_in[15];
  const float* out_b   = (const float*)d_in[16];
  const float* bn1     = (const float*)d_in[17];
  const float* bn2     = (const float*)d_in[18];
  float* out = (float*)d_out;

  // ---- workspace bump allocator -------------------------------------------
  char* ws = (char*)d_ws;
  size_t off = 0;
  auto alloc = [&](size_t bytes) -> void* {
    void* p = ws + off;
    off += (bytes + 255) & ~(size_t)255;
    return p;
  };
  _Float16* x16    = (_Float16*)alloc((size_t)BB * LL * EPAD * 2);
  _Float16* Wih16  = (_Float16*)alloc((size_t)G3H * EPAD * 2);
  _Float16* Whh16  = (_Float16*)alloc((size_t)G3H * HH * 2);
  _Float16* dWih16 = (_Float16*)alloc((size_t)G3H * HH * 2);
  _Float16* dWhh16 = (_Float16*)alloc((size_t)G3H * HH * 2);
  _Float16* comb16 = (_Float16*)alloc((size_t)HH * CAT2 * 2);
  _Float16* outW16 = (_Float16*)alloc((size_t)TT * HH * 2);
  float*    gi     = (float*)   alloc((size_t)BB * G3H * 4);
  float*    gh     = (float*)   alloc((size_t)BB * G3H * 4);
  float*    h      = (float*)   alloc((size_t)BB * HH * 4);
  _Float16* h16    = (_Float16*)alloc((size_t)BB * HH * 2);
  float*    encout = (float*)   alloc((size_t)BB * LL * HH * 4);
  float*    emb32  = (float*)   alloc((size_t)BB * DD * 4);
  _Float16* cat2   = (_Float16*)alloc((size_t)BB * CAT2 * 2);
  _Float16* o16    = (_Float16*)alloc((size_t)BB * HH * 2);
  _Float16* bnh16  = (_Float16*)alloc((size_t)BB * HH * 2);
  int*      ninp   = (int*)     alloc((size_t)BB * 4);

  auto cvt = [&](const float* s, _Float16* d, int rows, int cin, int cout) {
    long total = (long)rows * cout;
    cvt_f16_pad<<<dim3((unsigned)((total + 255) / 256)), 256, 0, stream>>>(s, d, rows, cin, cout);
  };
  auto gemm = [&](const _Float16* A, long lda, const _Float16* W, long ldb,
                  const float* bias, float* C, long ldc, _Float16* Ch, long ldch,
                  int M, int N, int K, const float* bnp, int relu) {
    dim3 g((unsigned)(N / 64), (unsigned)(M / 128));
    gemm_wmma_f16<<<g, 256, 0, stream>>>(A, lda, W, ldb, bias, C, ldc, Ch, ldch,
                                         M, N, K, bnp, relu);
  };

  // ---- one-time conversions (deterministic, re-run every call) ------------
  cvt(encWih, Wih16,  G3H, EE, EPAD);
  cvt(encWhh, Whh16,  G3H, HH, HH);
  cvt(decWih, dWih16, G3H, HH, HH);
  cvt(decWhh, dWhh16, G3H, HH, HH);
  cvt(comb_W, comb16, HH,  CAT2, CAT2);
  cvt(out_W,  outW16, TT,  HH, HH);

  // ---- encoder -------------------------------------------------------------
  enc_embed<<<dim3(BB * LL), 256, 0, stream>>>(tokens, w2v, x16);
  hipMemsetAsync(h,   0, (size_t)BB * HH * 4, stream);
  hipMemsetAsync(h16, 0, (size_t)BB * HH * 2, stream);

  for (int t = 0; t < LL; ++t) {
    // gi_t = x[:,t,:] @ Wih^T + bih   (rows strided by L*EPAD)
    gemm(x16 + (long)t * EPAD, (long)LL * EPAD, Wih16, EPAD, enc_bih,
         gi, G3H, nullptr, 0, BB, G3H, EPAD, nullptr, 0);
    // gh_t = h @ Whh^T + bhh
    gemm(h16, HH, Whh16, HH, enc_bhh, gh, G3H, nullptr, 0, BB, G3H, HH, nullptr, 0);
    gru_elem<<<dim3(HH / 256, BB), 256, 0, stream>>>(
        gi, gh, h, h16, encout + (long)t * HH, nullptr, nullptr);
  }

  // ---- decoder -------------------------------------------------------------
  fill_int<<<dim3(2), 256, 0, stream>>>(ninp, TT, BB);   // SOS token = T
  for (int step = 0; step < LL; ++step) {
    dec_embed<<<dim3(BB), 256, 0, stream>>>(ninp, dec_emb, emb32, cat2);
    attn_kernel<<<dim3(BB), 256, 0, stream>>>(emb32, h, attn_W, attn_b,
                                              encout, cat2 + DD);
    // o = relu(bn2(cat2 @ comb_W^T + comb_b)) -> f16
    gemm(cat2, CAT2, comb16, CAT2, comb_b, nullptr, 0, o16, HH,
         BB, HH, CAT2, bn2, 1);
    // gi = o @ dec_Wih^T + bih ; gh = h @ dec_Whh^T + bhh
    gemm(o16, HH, dWih16, HH, dec_bih, gi, G3H, nullptr, 0, BB, G3H, HH, nullptr, 0);
    gemm(h16, HH, dWhh16, HH, dec_bhh, gh, G3H, nullptr, 0, BB, G3H, HH, nullptr, 0);
    gru_elem<<<dim3(HH / 256, BB), 256, 0, stream>>>(
        gi, gh, h, h16, nullptr, bnh16, bn1);
    // logits -> directly into d_out rows b*L + step (strided)
    gemm(bnh16, HH, outW16, HH, out_b, out + (long)step * TT, (long)LL * TT,
         nullptr, 0, BB, TT, HH, nullptr, 0);
    lsm_argmax<<<dim3(BB), TT, 0, stream>>>(out + (long)step * TT, (long)LL * TT, ninp);
  }
}
